// FusionModel_33071248179270
// MI455X (gfx1250) — matrix operand order repaired and checked by reading, not compiled, not run on hardware
//
#include <hip/hip_runtime.h>
#include <math.h>

// ---------------- problem constants ----------------
#define BB   256
#define GG   186
#define FF   265
#define NN   (BB*GG)        // 47616 nodes per band
#define EE   1523712        // edges per band
#define NBANDS 5
#define HHDS 4              // GAT heads
#define CC   24             // GAT out channels per head
#define HC   (HHDS*CC)      // 96
#define EMB  125
#define SEQ  40             // 5 bands * 8 conv channels
#define AH   5
#define HD   25

// ---------------- WMMA types ----------------
typedef __attribute__((ext_vector_type(16))) __bf16        v16bf;
typedef __attribute__((ext_vector_type(8)))  float         v8f;
typedef __attribute__((ext_vector_type(8)))  unsigned int  v8u;

union FragU { v8u u; v16bf b; };

__device__ __forceinline__ unsigned short f2bf(float f) {
    unsigned int u = __float_as_uint(f);
    unsigned int r = (u + 0x7FFFu + ((u >> 16) & 1u)) >> 16;
    return (unsigned short)r;
}

// ---------------- optional CDNA5 async LDS copy path ----------------
#if defined(__gfx1250__) && __has_builtin(__builtin_amdgcn_global_load_async_to_lds_b128) && __has_builtin(__builtin_amdgcn_s_wait_asynccnt)
#define HAVE_ASYNC_LDS 1
typedef int v4i_gcc __attribute__((vector_size(4 * sizeof(int))));
__device__ __forceinline__ void async_b128(const void* g, void* l) {
    __builtin_amdgcn_global_load_async_to_lds_b128(
        (__attribute__((address_space(1))) v4i_gcc*)(uintptr_t)g,
        (__attribute__((address_space(3))) v4i_gcc*)(uintptr_t)l,
        0, 0);
}
#else
#define HAVE_ASYNC_LDS 0
#endif

// =====================================================================
// Generic GEMM:  Cout[M x Ncols] = act( A[M x K] (opt. per-k affine) * W[Ncols x K]^T + bias )
// bf16 WMMA 16x16x32, block = 128 threads = 4 waves; per block: 64 rows x 96 cols.
// =====================================================================
#define SA_STR 40
#define SW_STR 40

__global__ void __launch_bounds__(128)
wmma_gemm_nt(const float* __restrict__ A, int lda,
             const float* __restrict__ scale, const float* __restrict__ shift, float scaleMul,
             const float* __restrict__ W, const float* __restrict__ bias,
             float* __restrict__ Cout, int ldc,
             int M, int Ncols, int K, float slope)
{
    __shared__ unsigned short sA[64 * SA_STR];
    __shared__ unsigned short sW[96 * SW_STR];

    const int tid  = threadIdx.x;
    const int lane = tid & 31;
    const int wave = tid >> 5;
    const int mrow = lane & 15;
    const int half = lane >> 4;

    const int rowBlk = blockIdx.x * 64;
    const int colBlk = blockIdx.y * 96;

    v8f zero = {0.f,0.f,0.f,0.f,0.f,0.f,0.f,0.f};
    v8f acc[6];
#pragma unroll
    for (int ct = 0; ct < 6; ++ct) acc[ct] = zero;

    const int ksteps = (K + 31) >> 5;
    for (int ks = 0; ks < ksteps; ++ks) {
        const int k0 = ks << 5;
        // stage A tile (64x32) with fused per-feature affine + f32->bf16
        if (scale) {
            for (int idx = tid; idx < 64 * 32; idx += 128) {
                int r = idx >> 5, kk = idx & 31;
                int gr = rowBlk + r, gk = k0 + kk;
                float v = 0.f;
                if (gr < M && gk < K)
                    v = A[(size_t)gr * lda + gk] * scale[gk] * scaleMul + shift[gk];
                sA[r * SA_STR + kk] = f2bf(v);
            }
        } else {
            for (int idx = tid; idx < 64 * 32; idx += 128) {
                int r = idx >> 5, kk = idx & 31;
                int gr = rowBlk + r, gk = k0 + kk;
                float v = 0.f;
                if (gr < M && gk < K) v = A[(size_t)gr * lda + gk];
                sA[r * SA_STR + kk] = f2bf(v);
            }
        }
        // stage W panel (96x32)
        for (int idx = tid; idx < 96 * 32; idx += 128) {
            int r = idx >> 5, kk = idx & 31;
            int gc = colBlk + r, gk = k0 + kk;
            float v = 0.f;
            if (gc < Ncols && gk < K) v = W[(size_t)gc * K + gk];
            sW[r * SW_STR + kk] = f2bf(v);
        }
        // prefetch next K-step tiles (global_prefetch_b8; one line per row)
        if (ks + 1 < ksteps) {
            int nk = k0 + 32;
            if (nk < K) {
                if (tid < 64) {
                    int gr = rowBlk + tid;
                    if (gr < M) __builtin_prefetch(&A[(size_t)gr * lda + nk], 0, 1);
                } else {
                    int r = tid - 64;
                    int gc = colBlk + r;
                    if (gc < Ncols) __builtin_prefetch(&W[(size_t)gc * K + nk], 0, 1);
                    gc = colBlk + r + 64;
                    if (r < 32 && gc < Ncols) __builtin_prefetch(&W[(size_t)gc * K + nk], 0, 1);
                }
            }
        }
        __syncthreads();

        // A fragment: ISA 16-bit A 16x32 layout
        FragU af;
        const unsigned short* arow = &sA[(wave * 16 + mrow) * SA_STR];
#pragma unroll
        for (int j = 0; j < 8; ++j) {
            int k = (j < 4) ? (half * 8 + 2 * j) : (16 + half * 8 + 2 * (j - 4));
            af.u[j] = *(const unsigned int*)(arow + k);
        }
#pragma unroll
        for (int ct = 0; ct < 6; ++ct) {
            // B fragment: ISA 16-bit B 32x16 layout (same K order as A fragment)
            FragU bf;
            const unsigned short* brow = &sW[(ct * 16 + mrow) * SW_STR];
#pragma unroll
            for (int j = 0; j < 8; ++j) {
                int k = (j < 4) ? (half * 8 + 2 * j) : (16 + half * 8 + 2 * (j - 4));
                bf.u[j] = *(const unsigned int*)(brow + k);
            }
            acc[ct] = __builtin_amdgcn_wmma_f32_16x16x32_bf16(
                false, af.b, false, bf.b, (short)0, acc[ct], false, false);
        }
        __syncthreads();
    }

    // epilogue: C/D layout: VGPR r -> row r (+8 for upper half), col = lane&15
    const int rowBase = rowBlk + wave * 16 + half * 8;
#pragma unroll
    for (int ct = 0; ct < 6; ++ct) {
        int col = colBlk + ct * 16 + mrow;
        if (col >= Ncols) continue;
        float bv = bias ? bias[col] : 0.f;
#pragma unroll
        for (int r = 0; r < 8; ++r) {
            int row = rowBase + r;
            if (row < M) {
                float v = acc[ct][r] + bv;
                v = v > 0.f ? v : v * slope;     // slope==1 -> identity
                Cout[(size_t)row * ldc + col] = v;
            }
        }
    }
}

// =====================================================================
// GAT kernels
// =====================================================================
__device__ __forceinline__ void atomicMaxF(float* addr, float val) {
    int old = __float_as_int(*addr);
    while (val > __int_as_float(old)) {
        int prev = atomicCAS((int*)addr, old, __float_as_int(val));
        if (prev == old) break;
        old = prev;
    }
}

__global__ void gat_coef_kernel(const float* __restrict__ h,
                                const float* __restrict__ att_s, const float* __restrict__ att_d,
                                float* __restrict__ a_s, float* __restrict__ a_d,
                                float* __restrict__ m)
{
    int n = blockIdx.x * blockDim.x + threadIdx.x;
    if (n >= NN) return;
    const float* hr = h + (size_t)n * HC;
#pragma unroll
    for (int hh = 0; hh < HHDS; ++hh) {
        float ss = 0.f, dd = 0.f;
#pragma unroll
        for (int c = 0; c < CC; ++c) {
            float v = hr[hh * CC + c];
            ss += v * att_s[hh * CC + c];
            dd += v * att_d[hh * CC + c];
        }
        a_s[n * 4 + hh] = ss;
        a_d[n * 4 + hh] = dd;
        float a = ss + dd;
        a = a > 0.f ? a : 0.2f * a;
        m[n * 4 + hh] = a;              // self-loop initializes segment max
    }
}

__global__ void edge_max_kernel(const int* __restrict__ src, const int* __restrict__ dst,
                                const float* __restrict__ a_s, const float* __restrict__ a_d,
                                float* __restrict__ m)
{
    int e = blockIdx.x * blockDim.x + threadIdx.x;
    if (e >= EE) return;
    int s = src[e], d = dst[e];
#pragma unroll
    for (int hh = 0; hh < HHDS; ++hh) {
        float a = a_s[s * 4 + hh] + a_d[d * 4 + hh];
        a = a > 0.f ? a : 0.2f * a;
        atomicMaxF(&m[d * 4 + hh], a);
    }
}

__global__ void denom_init_kernel(const float* __restrict__ a_s, const float* __restrict__ a_d,
                                  const float* __restrict__ m, float* __restrict__ denom)
{
    int n = blockIdx.x * blockDim.x + threadIdx.x;
    if (n >= NN) return;
#pragma unroll
    for (int hh = 0; hh < HHDS; ++hh) {
        float a = a_s[n * 4 + hh] + a_d[n * 4 + hh];
        a = a > 0.f ? a : 0.2f * a;
        denom[n * 4 + hh] = expf(a - m[n * 4 + hh]);   // self-loop term
    }
}

__global__ void edge_denom_kernel(const int* __restrict__ src, const int* __restrict__ dst,
                                  const float* __restrict__ a_s, const float* __restrict__ a_d,
                                  const float* __restrict__ m, float* __restrict__ denom)
{
    int e = blockIdx.x * blockDim.x + threadIdx.x;
    if (e >= EE) return;
    int s = src[e], d = dst[e];
#pragma unroll
    for (int hh = 0; hh < HHDS; ++hh) {
        float a = a_s[s * 4 + hh] + a_d[d * 4 + hh];
        a = a > 0.f ? a : 0.2f * a;
        atomicAdd(&denom[d * 4 + hh], expf(a - m[d * 4 + hh]));
    }
}

// one wave per edge-entry (edges then self loops); lane covers 3 of 96 values
__global__ void gat_agg_kernel(const int* __restrict__ src, const int* __restrict__ dst,
                               const float* __restrict__ h,
                               const float* __restrict__ a_s, const float* __restrict__ a_d,
                               const float* __restrict__ m, const float* __restrict__ denom,
                               float* __restrict__ outp)
{
    int wid  = (blockIdx.x * blockDim.x + threadIdx.x) >> 5;
    int lane = threadIdx.x & 31;
    if (wid >= EE + NN) return;
    int s, d;
    if (wid < EE) { s = src[wid]; d = dst[wid]; }
    else          { s = d = wid - EE; }
#pragma unroll
    for (int c = 0; c < 3; ++c) {
        int j  = lane + 32 * c;       // 0..95
        int hh = j / CC;
        float a = a_s[s * 4 + hh] + a_d[d * 4 + hh];
        a = a > 0.f ? a : 0.2f * a;
        float alpha = expf(a - m[d * 4 + hh]) / (denom[d * 4 + hh] + 1e-16f);
        atomicAdd(&outp[(size_t)d * HC + j], alpha * h[(size_t)s * HC + j]);
    }
}

// mean over heads + bias + ELU, write transposed (B, C, G) for conv
__global__ void elu_kernel(const float* __restrict__ gat_out, const float* __restrict__ gbias,
                           float* __restrict__ convin)
{
    int idx = blockIdx.x * blockDim.x + threadIdx.x;
    if (idx >= NN * CC) return;
    int n = idx / CC, c = idx % CC;
    const float* r = gat_out + (size_t)n * HC;
    float v = 0.25f * (r[c] + r[CC + c] + r[2 * CC + c] + r[3 * CC + c]) + gbias[c];
    v = v > 0.f ? v : expm1f(v);
    int b = n / GG, g = n % GG;
    convin[((size_t)b * CC + c) * GG + g] = v;
}

// =====================================================================
// Conv1d VALID (24 -> 8 channels, kernel 62): block per batch element
// LDS staging uses the CDNA5 async global->LDS path when available.
// =====================================================================
__global__ void __launch_bounds__(256)
conv_kernel(const float* __restrict__ convin, const float* __restrict__ cw,
            const float* __restrict__ cb, float* __restrict__ xc, int band)
{
    alignas(16) __shared__ float sin_s[CC * GG];        // 24*186  = 17856 B
    alignas(16) __shared__ float sw_s[8 * CC * 62];     // 8*24*62 = 47616 B
    int b = blockIdx.x;
#if HAVE_ASYNC_LDS
    {
        const char* gi = (const char*)(convin + (size_t)b * CC * GG);
        for (int i = threadIdx.x; i < (CC * GG * 4) / 16; i += 256)
            async_b128(gi + i * 16, (char*)sin_s + i * 16);
        const char* gw = (const char*)cw;
        for (int i = threadIdx.x; i < (8 * CC * 62 * 4) / 16; i += 256)
            async_b128(gw + i * 16, (char*)sw_s + i * 16);
        __builtin_amdgcn_s_wait_asynccnt(0);
    }
#else
    for (int i = threadIdx.x; i < CC * GG; i += 256) sin_s[i] = convin[(size_t)b * CC * GG + i];
    for (int i = threadIdx.x; i < 8 * CC * 62; i += 256) sw_s[i] = cw[i];
#endif
    __syncthreads();
    for (int idx = threadIdx.x; idx < 8 * 125; idx += 256) {
        int o = idx / 125, t = idx % 125;
        float acc = cb[o];
        for (int c = 0; c < CC; ++c) {
            const float* ip = &sin_s[c * GG + t];
            const float* wp = &sw_s[(o * CC + c) * 62];
#pragma unroll 2
            for (int k = 0; k < 62; ++k) acc += ip[k] * wp[k];
        }
        acc = acc > 0.f ? acc : 0.01f * acc;
        xc[((size_t)b * SEQ + band * 8 + o) * EMB + t] = acc;
    }
}

// =====================================================================
// LayerNorm over last dim (125); one wave per row
// =====================================================================
__global__ void ln_kernel(const float* __restrict__ xc, const float* __restrict__ g,
                          const float* __restrict__ be, float* __restrict__ y)
{
    int row = blockIdx.x * (blockDim.x >> 5) + (threadIdx.x >> 5);
    if (row >= BB * SEQ) return;
    int lane = threadIdx.x & 31;
    const float* xr = xc + (size_t)row * EMB;
    float s = 0.f, s2 = 0.f;
    for (int i = lane; i < EMB; i += 32) { float v = xr[i]; s += v; s2 += v * v; }
#pragma unroll
    for (int off = 16; off > 0; off >>= 1) {
        s  += __shfl_xor(s,  off, 32);
        s2 += __shfl_xor(s2, off, 32);
    }
    float mu  = s * (1.f / EMB);
    float var = s2 * (1.f / EMB) - mu * mu;
    float rstd = rsqrtf(var + 1e-5f);
    float* yr = y + (size_t)row * EMB;
    for (int i = lane; i < EMB; i += 32) yr[i] = (xr[i] - mu) * rstd * g[i] + be[i];
}

// =====================================================================
// Self-attention (per batch, per head): scores + softmax + apply, seq=40, hd=25
// =====================================================================
__global__ void __launch_bounds__(256)
attn_kernel(const float* __restrict__ qkv, float* __restrict__ o)
{
    __shared__ float qs[SEQ * HD], ks_[SEQ * HD], vs[SEQ * HD], sc[SEQ * SEQ];
    int b = blockIdx.x, h = blockIdx.y;
    for (int idx = threadIdx.x; idx < SEQ * HD; idx += 256) {
        int s = idx / HD, d = idx % HD;
        const float* base = qkv + ((size_t)(b * SEQ + s)) * (3 * EMB) + h * HD + d;
        qs[idx]  = base[0];
        ks_[idx] = base[EMB];
        vs[idx]  = base[2 * EMB];
    }
    __syncthreads();
    for (int idx = threadIdx.x; idx < SEQ * SEQ; idx += 256) {
        int qi = idx / SEQ, ki = idx % SEQ;
        float acc = 0.f;
#pragma unroll
        for (int d = 0; d < HD; ++d) acc += qs[qi * HD + d] * ks_[ki * HD + d];
        sc[idx] = acc * 0.2f;                      // 1/sqrt(25)
    }
    __syncthreads();
    if (threadIdx.x < SEQ) {
        int qi = threadIdx.x;
        float mx = -3.4e38f;
        for (int k = 0; k < SEQ; ++k) mx = fmaxf(mx, sc[qi * SEQ + k]);
        float sum = 0.f;
        for (int k = 0; k < SEQ; ++k) { float e = expf(sc[qi * SEQ + k] - mx); sc[qi * SEQ + k] = e; sum += e; }
        float inv = 1.f / sum;
        for (int k = 0; k < SEQ; ++k) sc[qi * SEQ + k] *= inv;
    }
    __syncthreads();
    for (int idx = threadIdx.x; idx < SEQ * HD; idx += 256) {
        int qi = idx / HD, d = idx % HD;
        float acc = 0.f;
#pragma unroll
        for (int k = 0; k < SEQ; ++k) acc += sc[qi * SEQ + k] * vs[k * HD + d];
        o[((size_t)(b * SEQ + qi)) * EMB + h * HD + d] = acc;
    }
}

// =====================================================================
// host launcher
// =====================================================================
extern "C" void kernel_launch(void* const* d_in, const int* in_sizes, int n_in,
                              void* d_out, int out_size, void* d_ws, size_t ws_size,
                              hipStream_t stream) {
    (void)in_sizes; (void)n_in; (void)out_size; (void)ws_size;
    const float* x_bands  = (const float*)d_in[0];
    const int*   edges    = (const int*)  d_in[1];
    const float* bn_gamma = (const float*)d_in[2];
    const float* bn_beta  = (const float*)d_in[3];
    const float* lin_w    = (const float*)d_in[4];
    const float* att_src  = (const float*)d_in[5];
    const float* att_dst  = (const float*)d_in[6];
    const float* gat_bias = (const float*)d_in[7];
    const float* conv_w   = (const float*)d_in[8];
    const float* conv_b   = (const float*)d_in[9];
    const float* ln_g     = (const float*)d_in[10];
    const float* ln_b     = (const float*)d_in[11];
    const float* inp_w    = (const float*)d_in[12];
    const float* inp_b    = (const float*)d_in[13];
    const float* outp_w   = (const float*)d_in[14];
    const float* outp_b   = (const float*)d_in[15];
    const float* fus_w    = (const float*)d_in[16];
    const float* fus_b    = (const float*)d_in[17];
    float* out = (float*)d_out;

    float* ws = (float*)d_ws;
    float* xc      = ws;                                  // B*40*125
    float* hbuf    = xc + (size_t)BB * SEQ * EMB;         // N*96
    float* gat_out = hbuf + (size_t)NN * HC;              // N*96
    float* a_s     = gat_out + (size_t)NN * HC;           // N*4
    float* a_d     = a_s + (size_t)NN * 4;
    float* mbuf    = a_d + (size_t)NN * 4;
    float* denom   = mbuf + (size_t)NN * 4;
    float* convin  = denom + (size_t)NN * 4;              // B*24*186
    // attention phase overlays the (finished) band scratch:
    float* ybuf  = hbuf;
    float* obuf  = hbuf + (size_t)BB * SEQ * EMB;
    float* oproj = obuf + (size_t)BB * SEQ * EMB;
    float* qkv   = gat_out;                               // 10240*375 fits in N*96

    const float scaleMul = 1.0f / sqrtf(1.0f + 1e-5f);

    for (int band = 0; band < NBANDS; ++band) {
        const float* x   = x_bands + (size_t)band * NN * FF;
        const int*   src = edges + (size_t)band * 2 * EE;
        const int*   dst = src + EE;
        const float* lw  = lin_w + (size_t)band * HC * FF;

        // h = BN(x) @ lin_w^T  (WMMA, BN fused into A load)
        wmma_gemm_nt<<<dim3((NN + 63) / 64, 1), 128, 0, stream>>>(
            x, FF, bn_gamma + band * FF, bn_beta + band * FF, scaleMul,
            lw, nullptr, hbuf, HC, NN, HC, FF, 1.0f);

        gat_coef_kernel<<<(NN + 255) / 256, 256, 0, stream>>>(
            hbuf, att_src + band * HC, att_dst + band * HC, a_s, a_d, mbuf);
        edge_max_kernel<<<(EE + 255) / 256, 256, 0, stream>>>(src, dst, a_s, a_d, mbuf);
        denom_init_kernel<<<(NN + 255) / 256, 256, 0, stream>>>(a_s, a_d, mbuf, denom);
        edge_denom_kernel<<<(EE + 255) / 256, 256, 0, stream>>>(src, dst, a_s, a_d, mbuf, denom);

        (void)hipMemsetAsync(gat_out, 0, (size_t)NN * HC * sizeof(float), stream);
        {
            long long waves = (long long)EE + NN;          // one wave per edge-entry
            long long blocks = (waves + 7) / 8;            // 256 thr = 8 waves
            gat_agg_kernel<<<(unsigned)blocks, 256, 0, stream>>>(
                src, dst, hbuf, a_s, a_d, mbuf, denom, gat_out);
        }
        elu_kernel<<<(NN * CC + 255) / 256, 256, 0, stream>>>(
            gat_out, gat_bias + band * CC, convin);
        conv_kernel<<<BB, 256, 0, stream>>>(
            convin, conv_w + (size_t)band * 8 * CC * 62, conv_b + band * 8, xc, band);
    }

    // LayerNorm
    ln_kernel<<<(BB * SEQ + 7) / 8, 256, 0, stream>>>(xc, ln_g, ln_b, ybuf);

    // qkv = y @ in_proj_w^T + b   (10240 x 375, K=125)
    wmma_gemm_nt<<<dim3((BB * SEQ + 63) / 64, (3 * EMB + 95) / 96), 128, 0, stream>>>(
        ybuf, EMB, nullptr, nullptr, 1.0f, inp_w, inp_b, qkv, 3 * EMB,
        BB * SEQ, 3 * EMB, EMB, 1.0f);

    // softmax attention
    attn_kernel<<<dim3(BB, AH), 256, 0, stream>>>(qkv, obuf);

    // out_proj (10240 x 125, K=125)
    wmma_gemm_nt<<<dim3((BB * SEQ + 63) / 64, (EMB + 95) / 96), 128, 0, stream>>>(
        obuf, EMB, nullptr, nullptr, 1.0f, outp_w, outp_b, oproj, EMB,
        BB * SEQ, EMB, EMB, 1.0f);

    // fusion head with fused leaky-ReLU: (256 x 3, K=5000)
    wmma_gemm_nt<<<dim3((BB + 63) / 64, 1), 128, 0, stream>>>(
        oproj, SEQ * EMB, nullptr, nullptr, 1.0f, fus_w, fus_b, out, 3,
        BB, 3, SEQ * EMB, 0.01f);
}
